// DAGEmbedding_84215718740063
// MI455X (gfx1250) — matrix operand order repaired and checked by reading, not compiled, not run on hardware
//
#include <hip/hip_runtime.h>
#include <hip/hip_bf16.h>

typedef __attribute__((ext_vector_type(16))) _Float16 v16h;
typedef __attribute__((ext_vector_type(8)))  _Float16 v8h;
typedef __attribute__((ext_vector_type(8)))  float    v8f;

#define MTILE 128      // edges per block
#define SFEAT 208      // LDS row stride in halves (192 + pad)

static __device__ __forceinline__ v8f wmma_f16(v16h a, v16h b, v8f c) {
  // D = A(16x32 f16) x B(32x16 f16) + C(16x16 f32)
  return __builtin_amdgcn_wmma_f32_16x16x32_f16(false, a, false, b, (short)0, c,
                                                false, false);
}

// A fragment (16x32 f16) from LDS feat tile. Layout per ISA 7.12.2:
// lanes 0-15: h0..7 = K 0..7, h8..15 = K 16..23 ; lanes 16-31: +8.
static __device__ __forceinline__ v16h load_a_lds(const _Float16* sF, int mbase,
                                                  int lane, int kk) {
  int row = mbase + (lane & 15);
  int hi  = lane >> 4;
  const _Float16* p = sF + row * SFEAT + kk + hi * 8;
  v8h lo = *(const v8h*)(p);
  v8h hp = *(const v8h*)(p + 16);
  v16h a;
#pragma unroll
  for (int j = 0; j < 8; ++j) { a[j] = lo[j]; a[8 + j] = hp[j]; }
  return a;
}

// B fragment (32x16 f16) from row-major weight [out][ld]:
// lane col n = lane&15; lanes 0-15: h0..7 = K 0..7, h8..15 = K 16..23 ; lanes 16-31: +8.
static __device__ __forceinline__ v16h load_b_gl(const _Float16* W, int ld,
                                                 int ntBase, int lane, int kk) {
  int col = lane & 15;
  int hi  = lane >> 4;
  const _Float16* p = W + (size_t)(ntBase + col) * ld + kk + hi * 8;
  v8h lo = *(const v8h*)(p);
  v8h hp = *(const v8h*)(p + 16);
  v16h b;
#pragma unroll
  for (int j = 0; j < 8; ++j) { b[j] = lo[j]; b[8 + j] = hp[j]; }
  return b;
}

// ---------------------------------------------------------------- utilities
__global__ void zero_f32(float* __restrict__ p, long n) {
  long i = (long)blockIdx.x * 256 + threadIdx.x;
  long stride = (long)gridDim.x * 256;
  for (; i < n; i += stride) p[i] = 0.f;
}

__global__ void count_kernel(const int* __restrict__ twi, float* __restrict__ cnt,
                             int E, int Nn) {
  long idx = (long)blockIdx.x * 256 + threadIdx.x;
  if (idx < 3L * E) {
    int comp = (int)(idx / E);
    int node = twi[idx];
    atomicAdd(&cnt[(size_t)comp * Nn + node], 1.0f);
  }
}

__global__ void wconv(const float* __restrict__ WT, const float* __restrict__ WM,
                      const float* __restrict__ WB, const float* __restrict__ Wtw,
                      _Float16* __restrict__ Wcat, _Float16* __restrict__ Wtwh,
                      int layer) {
  int idx = blockIdx.x * 256 + threadIdx.x;
  if (idx < 192 * 192) {
    int o = idx / 192, k = idx - o * 192;
    const float* src = (o < 64) ? WT : (o < 128 ? WM : WB);
    Wcat[idx] = (_Float16)src[(size_t)layer * 64 * 192 + (size_t)(o & 63) * 192 + k];
  } else if (idx < 192 * 192 + 64 * 64) {
    int j = idx - 192 * 192;
    Wtwh[j] = (_Float16)Wtw[(size_t)layer * 4096 + j];
  }
}

// ------------------------------------------------- edge pass 1: GEMM + stats
__global__ __launch_bounds__(256)
void edge_pass1(const float* __restrict__ x, const int* __restrict__ twi,
                const _Float16* __restrict__ Wcat, float* __restrict__ estats,
                int E, int Nn) {
  __shared__ __align__(32) _Float16 sF[MTILE * SFEAT];
  __shared__ int   sIdx[3 * MTILE];
  __shared__ float sStat[384];
  const int t  = threadIdx.x;
  const int eb = blockIdx.x * MTILE;
  if (t < 384) sStat[t] = 0.f;
  for (int k = t; k < 3 * MTILE; k += 256) {
    int comp = k >> 7, e = k & (MTILE - 1);
    int ge = eb + e;
    sIdx[k] = (ge < E) ? twi[(size_t)comp * E + ge] : -1;
  }
  __syncthreads();
  for (int task = t; task < 3 * MTILE * 4; task += 256) {
    int part = task & 3, row = task >> 2;
    int comp = row >> 7, e = row & (MTILE - 1);
    int node = sIdx[comp * MTILE + e];
    _Float16* dst = &sF[e * SFEAT + comp * 64 + part * 16];
    if (node >= 0) {
      const float4* src = (const float4*)(x + (size_t)node * 64 + part * 16);
      float4 f0 = src[0], f1 = src[1], f2 = src[2], f3 = src[3];
      v8h a, b;
      a[0]=(_Float16)f0.x; a[1]=(_Float16)f0.y; a[2]=(_Float16)f0.z; a[3]=(_Float16)f0.w;
      a[4]=(_Float16)f1.x; a[5]=(_Float16)f1.y; a[6]=(_Float16)f1.z; a[7]=(_Float16)f1.w;
      b[0]=(_Float16)f2.x; b[1]=(_Float16)f2.y; b[2]=(_Float16)f2.z; b[3]=(_Float16)f2.w;
      b[4]=(_Float16)f3.x; b[5]=(_Float16)f3.y; b[6]=(_Float16)f3.z; b[7]=(_Float16)f3.w;
      *(v8h*)dst = a; *(v8h*)(dst + 8) = b;
    } else {
      v8h z = {};
      *(v8h*)dst = z; *(v8h*)(dst + 8) = z;
    }
  }
  __syncthreads();
  const int wave = t >> 5, lane = t & 31;
  const int mbase = wave * 16;
  v16h afr[6];
#pragma unroll
  for (int ks = 0; ks < 6; ++ks) afr[ks] = load_a_lds(sF, mbase, lane, ks * 32);
  for (int nt = 0; nt < 12; ++nt) {
    v8f c = {};
#pragma unroll
    for (int ks = 0; ks < 6; ++ks) {
      v16h b = load_b_gl(Wcat, 192, nt * 16, lane, ks * 32);
      c = wmma_f16(afr[ks], b, c);
    }
    float s = 0.f, q = 0.f;
#pragma unroll
    for (int r = 0; r < 8; ++r) { float v = c[r]; s += v; q += v * v; }
    int ch = nt * 16 + (lane & 15);
    atomicAdd(&sStat[ch], s);
    atomicAdd(&sStat[192 + ch], q);
  }
  __syncthreads();
  if (t < 384) atomicAdd(&estats[t], sStat[t]);
}

__global__ void fin_edge(const float* gT, const float* gM, const float* gB,
                         const float* bT, const float* bM, const float* bB,
                         const float* __restrict__ estats, float* __restrict__ esc,
                         int layer, float invE) {
  int ch = threadIdx.x;
  if (ch >= 192) return;
  int comp = ch >> 6, cl = ch & 63;
  const float* g  = (comp == 0) ? gT : (comp == 1 ? gM : gB);
  const float* be = (comp == 0) ? bT : (comp == 1 ? bM : bB);
  float gg = g[layer * 64 + cl], bb = be[layer * 64 + cl];
  float mu  = estats[ch] * invE;
  float var = estats[192 + ch] * invE - mu * mu;
  float sc  = gg * rsqrtf(var + 1e-5f);
  esc[ch]       = sc;
  esc[192 + ch] = bb - mu * sc;
}

// ---------------------------------- edge pass 2: GEMM + BN + ReLU + scatter
__global__ __launch_bounds__(256)
void edge_pass2(const float* __restrict__ x, const int* __restrict__ twi,
                const _Float16* __restrict__ Wcat, const float* __restrict__ esc,
                const float* __restrict__ cnt, float* __restrict__ macc,
                int E, int Nn) {
  __shared__ __align__(32) _Float16 sF[MTILE * SFEAT];
  __shared__ int   sIdx[3 * MTILE];
  __shared__ float sSc[384];
  const int t  = threadIdx.x;
  const int eb = blockIdx.x * MTILE;
  if (t < 384) sSc[t] = esc[t];
  for (int k = t; k < 3 * MTILE; k += 256) {
    int comp = k >> 7, e = k & (MTILE - 1);
    int ge = eb + e;
    sIdx[k] = (ge < E) ? twi[(size_t)comp * E + ge] : -1;
  }
  __syncthreads();
  for (int task = t; task < 3 * MTILE * 4; task += 256) {
    int part = task & 3, row = task >> 2;
    int comp = row >> 7, e = row & (MTILE - 1);
    int node = sIdx[comp * MTILE + e];
    _Float16* dst = &sF[e * SFEAT + comp * 64 + part * 16];
    if (node >= 0) {
      const float4* src = (const float4*)(x + (size_t)node * 64 + part * 16);
      float4 f0 = src[0], f1 = src[1], f2 = src[2], f3 = src[3];
      v8h a, b;
      a[0]=(_Float16)f0.x; a[1]=(_Float16)f0.y; a[2]=(_Float16)f0.z; a[3]=(_Float16)f0.w;
      a[4]=(_Float16)f1.x; a[5]=(_Float16)f1.y; a[6]=(_Float16)f1.z; a[7]=(_Float16)f1.w;
      b[0]=(_Float16)f2.x; b[1]=(_Float16)f2.y; b[2]=(_Float16)f2.z; b[3]=(_Float16)f2.w;
      b[4]=(_Float16)f3.x; b[5]=(_Float16)f3.y; b[6]=(_Float16)f3.z; b[7]=(_Float16)f3.w;
      *(v8h*)dst = a; *(v8h*)(dst + 8) = b;
    } else {
      v8h z = {};
      *(v8h*)dst = z; *(v8h*)(dst + 8) = z;
    }
  }
  __syncthreads();
  const int wave = t >> 5, lane = t & 31;
  const int mbase = wave * 16;
  const int hi = lane >> 4, ncol = lane & 15;
  v16h afr[6];
#pragma unroll
  for (int ks = 0; ks < 6; ++ks) afr[ks] = load_a_lds(sF, mbase, lane, ks * 32);
  for (int nt = 0; nt < 12; ++nt) {
    v8f c = {};
#pragma unroll
    for (int ks = 0; ks < 6; ++ks) {
      v16h b = load_b_gl(Wcat, 192, nt * 16, lane, ks * 32);
      c = wmma_f16(afr[ks], b, c);
    }
    int comp = nt >> 2;
    int chG  = nt * 16 + ncol;
    int chL  = (nt & 3) * 16 + ncol;
    float sc = sSc[chG], sh = sSc[192 + chG];
#pragma unroll
    for (int r = 0; r < 8; ++r) {
      int m = mbase + r + hi * 8;
      int node = sIdx[comp * MTILE + m];
      if (node >= 0) {
        float y = fmaxf(c[r] * sc + sh, 0.f);
        float rinv = 1.f / fmaxf(cnt[(size_t)comp * Nn + node], 1.f);
        atomicAdd(&macc[(size_t)node * 64 + chL], y * rinv);
      }
    }
  }
}

// ----------------------------------------------- node pass 1: GEMM + stats
__global__ __launch_bounds__(256)
void node_pass1(const float* __restrict__ macc, const _Float16* __restrict__ Wtwh,
                float* __restrict__ hpre, float* __restrict__ nstats, int Nn) {
  __shared__ float sStat[128];
  const int t = threadIdx.x;
  if (t < 128) sStat[t] = 0.f;
  __syncthreads();
  const int wave = t >> 5, lane = t & 31;
  const int nbase = blockIdx.x * 128 + wave * 16;
  const int hi = lane >> 4, ncol = lane & 15;
  const int rowNode = nbase + (lane & 15);
  const bool valid = rowNode < Nn;
  v16h afr[2];
#pragma unroll
  for (int ks = 0; ks < 2; ++ks) {
    v16h a = {};
    if (valid) {
      const float* p = macc + (size_t)rowNode * 64 + ks * 32 + hi * 8;
      float4 f0 = *(const float4*)(p);
      float4 f1 = *(const float4*)(p + 4);
      float4 f2 = *(const float4*)(p + 16);
      float4 f3 = *(const float4*)(p + 20);
      a[0]=(_Float16)f0.x; a[1]=(_Float16)f0.y; a[2]=(_Float16)f0.z; a[3]=(_Float16)f0.w;
      a[4]=(_Float16)f1.x; a[5]=(_Float16)f1.y; a[6]=(_Float16)f1.z; a[7]=(_Float16)f1.w;
      a[8]=(_Float16)f2.x; a[9]=(_Float16)f2.y; a[10]=(_Float16)f2.z; a[11]=(_Float16)f2.w;
      a[12]=(_Float16)f3.x; a[13]=(_Float16)f3.y; a[14]=(_Float16)f3.z; a[15]=(_Float16)f3.w;
    }
    afr[ks] = a;
  }
#pragma unroll
  for (int nt = 0; nt < 4; ++nt) {
    v8f c = {};
#pragma unroll
    for (int ks = 0; ks < 2; ++ks) {
      v16h b = load_b_gl(Wtwh, 64, nt * 16, lane, ks * 32);
      c = wmma_f16(afr[ks], b, c);
    }
    int ch = nt * 16 + ncol;
    float s = 0.f, q = 0.f;
#pragma unroll
    for (int r = 0; r < 8; ++r) {
      int node = nbase + r + hi * 8;
      if (node < Nn) hpre[(size_t)node * 64 + ch] = c[r];
      s += c[r]; q += c[r] * c[r];
    }
    atomicAdd(&sStat[ch], s);
    atomicAdd(&sStat[64 + ch], q);
  }
  __syncthreads();
  if (t < 128) atomicAdd(&nstats[t], sStat[t]);
}

__global__ void fin_node(const float* __restrict__ gtw, const float* __restrict__ btw,
                         const float* __restrict__ nstats, float* __restrict__ nsc,
                         int layer, float invN) {
  int ch = threadIdx.x;
  if (ch >= 64) return;
  float gg = gtw[layer * 64 + ch], bb = btw[layer * 64 + ch];
  float mu  = nstats[ch] * invN;
  float var = nstats[64 + ch] * invN - mu * mu;
  float sc  = gg * rsqrtf(var + 1e-5f);
  nsc[ch]      = sc;
  nsc[64 + ch] = bb - mu * sc;
}

// ----------------------------------------------- node pass 2: residual add
__global__ void node_pass2(const float* __restrict__ xin, const float* __restrict__ hpre,
                           const float* __restrict__ nsc, float* __restrict__ xout,
                           long total) {
  long idx = (long)blockIdx.x * 256 + threadIdx.x;
  if (idx < total) {
    int ch = (int)(idx & 63);
    float y = fmaxf(hpre[idx] * nsc[ch] + nsc[64 + ch], 0.f);
    xout[idx] = xin[idx] + y;
  }
}

static inline size_t alignUp(size_t v) { return (v + 255) & ~(size_t)255; }

extern "C" void kernel_launch(void* const* d_in, const int* in_sizes, int n_in,
                              void* d_out, int out_size, void* d_ws, size_t ws_size,
                              hipStream_t stream) {
  const float* x0     = (const float*)d_in[0];
  const int*   twi    = (const int*)  d_in[1];
  const float* WT     = (const float*)d_in[2];
  const float* gT     = (const float*)d_in[4];
  const float* betaT  = (const float*)d_in[5];
  const float* WM     = (const float*)d_in[6];
  const float* gM     = (const float*)d_in[8];
  const float* betaM  = (const float*)d_in[9];
  const float* WB     = (const float*)d_in[10];
  const float* gB     = (const float*)d_in[12];
  const float* betaB  = (const float*)d_in[13];
  const float* Wtw    = (const float*)d_in[14];
  const float* gtw    = (const float*)d_in[16];
  const float* betatw = (const float*)d_in[17];

  const int Nn = in_sizes[0] / 64;
  const int E  = in_sizes[1] / 3;
  float* out = (float*)d_out;

  char* ws = (char*)d_ws;
  size_t off = 0;
  float*    macc   = (float*)(ws + off); off = alignUp(off + (size_t)Nn * 64 * 4);
  float*    cnt    = (float*)(ws + off); off = alignUp(off + (size_t)3 * Nn * 4);
  float*    hpre   = (float*)(ws + off); off = alignUp(off + (size_t)Nn * 64 * 4);
  _Float16* Wcat   = (_Float16*)(ws + off); off = alignUp(off + 192 * 192 * 2);
  _Float16* Wtwh   = (_Float16*)(ws + off); off = alignUp(off + 64 * 64 * 2);
  float*    estats = (float*)(ws + off); off = alignUp(off + 384 * 4);
  float*    esc    = (float*)(ws + off); off = alignUp(off + 384 * 4);
  float*    nstats = (float*)(ws + off); off = alignUp(off + 128 * 4);
  float*    nsc    = (float*)(ws + off); off = alignUp(off + 128 * 4);

  const int edgeBlocks = (E + MTILE - 1) / MTILE;
  const int nodeBlocks = (Nn + 127) / 128;
  const long totF = (long)Nn * 64;

  // counts (fixed across layers)
  {
    long ncnt = (long)3 * Nn;
    zero_f32<<<(int)((ncnt + 255) / 256), 256, 0, stream>>>(cnt, ncnt);
    long nidx = (long)3 * E;
    count_kernel<<<(int)((nidx + 255) / 256), 256, 0, stream>>>(twi, cnt, E, Nn);
  }

  for (int layer = 0; layer < 2; ++layer) {
    const float* xcur = (layer == 0) ? x0 : out;
    zero_f32<<<(int)((totF + 255) / 256), 256, 0, stream>>>(macc, totF);
    zero_f32<<<1, 256, 0, stream>>>(estats, 384);
    zero_f32<<<1, 256, 0, stream>>>(nstats, 128);
    wconv<<<160, 256, 0, stream>>>(WT, WM, WB, Wtw, Wcat, Wtwh, layer);

    edge_pass1<<<edgeBlocks, 256, 0, stream>>>(xcur, twi, Wcat, estats, E, Nn);
    fin_edge<<<1, 192, 0, stream>>>(gT, gM, gB, betaT, betaM, betaB,
                                    estats, esc, layer, 1.0f / (float)E);
    edge_pass2<<<edgeBlocks, 256, 0, stream>>>(xcur, twi, Wcat, esc, cnt, macc, E, Nn);

    node_pass1<<<nodeBlocks, 256, 0, stream>>>(macc, Wtwh, hpre, nstats, Nn);
    fin_node<<<1, 64, 0, stream>>>(gtw, betatw, nstats, nsc, layer, 1.0f / (float)Nn);
    node_pass2<<<(int)((totF + 255) / 256), 256, 0, stream>>>(xcur, hpre, nsc, out, totF);
  }
}
